// MultiHeadAttention_46952582480196
// MI455X (gfx1250) — matrix operand order repaired and checked by reading, not compiled, not run on hardware
//
#include <hip/hip_runtime.h>

// ---------------------------------------------------------------------------
// Multi-head attention forward (B=4, S=2048, E=768, H=8, D=96), bf16 WMMA.
// cvt fp32->bf16 (weights transposed) -> TDM-fed TN GEMMs (Q,K,V) ->
// flash attention -> TDM-fed TN GEMM (out, fp32)
// ---------------------------------------------------------------------------

typedef __attribute__((ext_vector_type(16))) __bf16 v16bf;
typedef __attribute__((ext_vector_type(8)))  float  v8f;
typedef __attribute__((ext_vector_type(4)))  unsigned int u32x4;
typedef __attribute__((ext_vector_type(8)))  int i32x8;
typedef __attribute__((ext_vector_type(4)))  int i32x4;

#define EMB       768
#define HEADS     8
#define HEAD_DIM  96
#define SEQ       2048
#define BATCH     4

union FragU { uint4 u[2]; v16bf v; };

__device__ __forceinline__ v16bf ld_frag16(const __bf16* lo, const __bf16* hi) {
    FragU f;
    f.u[0] = *(const uint4*)lo;
    f.u[1] = *(const uint4*)hi;
    return f.v;
}

__device__ __forceinline__ v8f wmma_bf16(v16bf a, v16bf b, v8f c) {
    return __builtin_amdgcn_wmma_f32_16x16x32_bf16(false, a, false, b, (short)0, c,
                                                   false, false);
}

// Flat LDS pointer -> wave-relative LDS byte offset (low 32 bits of aperture addr)
__device__ __forceinline__ unsigned lds_off_u32(const void* p) {
    return (unsigned)(unsigned long long)p;
}

// ---------------------------------------------------------------------------
// TDM: async 2-D tile load (bf16 elements) global -> LDS with row padding.
// tile_d0 elems per contiguous row, tile_d1 rows, global row stride d0_stride
// elems. LDS pad: pad_amt+1 DWORDs after every 2^(pad_ivl+1) DWORDs of data.
// ---------------------------------------------------------------------------
__device__ __forceinline__ void tdm_load_tile_bf16(
        unsigned lds_byte_off, const void* gaddr,
        int tile_d0, int tile_d1, long d0_stride,
        int pad_ivl, int pad_amt)
{
    unsigned long long ga = (unsigned long long)gaddr;
    u32x4 g0;
    g0[0] = 1u;                                        // count=1, user D#
    g0[1] = lds_byte_off;                              // lds_addr
    g0[2] = (unsigned)(ga & 0xffffffffu);              // global_addr[31:0]
    g0[3] = (unsigned)((ga >> 32) & 0x01ffffffu)       // global_addr[56:32]
          | 0x80000000u;                               // type=2 (image)
    i32x8 g1;
    g1[0] = (1 << 16)                                  // data_size=1 -> 2 bytes
          | (1 << 20)                                  // pad_enable
          | (pad_ivl << 22) | (pad_amt << 25);
    g1[1] = (0xffff) << 16;                            // tensor_dim0 lo (huge, exact-fit)
    g1[2] = 0x7fff | (0xffff << 16);                   // tensor_dim0 hi | tensor_dim1 lo
    g1[3] = 0x7fff | (tile_d0 << 16);                  // tensor_dim1 hi | tile_dim0
    g1[4] = tile_d1 & 0xffff;                          // tile_dim1 | tile_dim2=0
    g1[5] = (int)(d0_stride & 0xffffffff);             // tensor_dim0_stride[31:0]
    g1[6] = (int)((d0_stride >> 32) & 0xffff);         // stride hi | dim1_stride lo = 0
    g1[7] = 0;
    i32x4 z4 = {0, 0, 0, 0};
#if defined(__clang_major__) && (__clang_major__ >= 23)
    i32x8 z8 = {0, 0, 0, 0, 0, 0, 0, 0};
    __builtin_amdgcn_tensor_load_to_lds(g0, g1, z4, z4, z8, 0);
#else
    __builtin_amdgcn_tensor_load_to_lds(g0, g1, z4, z4, 0);
#endif
}

// ---------------------------------------------------------------------------
// fp32 -> bf16 conversion (n multiple of 4)
// ---------------------------------------------------------------------------
__global__ __launch_bounds__(256)
void cvt_f32_to_bf16(const float* __restrict__ in, __bf16* __restrict__ out, int n) {
    int i = (blockIdx.x * 256 + threadIdx.x) * 4;
    if (i + 3 < n) {
        float4 f = *(const float4*)&in[i];
        out[i + 0] = (__bf16)f.x;
        out[i + 1] = (__bf16)f.y;
        out[i + 2] = (__bf16)f.z;
        out[i + 3] = (__bf16)f.w;
    }
}

// ---------------------------------------------------------------------------
// fp32 [rows][cols] -> bf16 transposed [cols][rows] (32x32 LDS tile transpose)
// ---------------------------------------------------------------------------
__global__ __launch_bounds__(256)
void cvt_transpose_bf16(const float* __restrict__ in, __bf16* __restrict__ out,
                        int rows, int cols) {
    __shared__ float tile[32][33];
    const int bx = blockIdx.x * 32, by = blockIdx.y * 32;
    const int tx = threadIdx.x & 31, ty = threadIdx.x >> 5;   // 32 x 8
    #pragma unroll
    for (int i = 0; i < 32; i += 8)
        tile[ty + i][tx] = in[(long)(by + ty + i) * cols + bx + tx];
    __syncthreads();
    #pragma unroll
    for (int i = 0; i < 32; i += 8)
        out[(long)(bx + ty + i) * rows + by + tx] = (__bf16)tile[tx][ty + i];
}

// ---------------------------------------------------------------------------
// TN GEMM: C[M,N] = A[M,K] * Bt[N,K]^T + bias[N]   (both operands K-contiguous)
// Block tile 128x128, K-step 64 (2 WMMA chunks per buffer), TDM double-buffered
// tiles, 8 waves = 2x4. LDS rows: 64 bf16 (128 B = 32 DW) + 16 B pad (4 DW)
// -> stride 72 elems = 144 B (multiple of 16 B; fragments stay b128 loads).
// ---------------------------------------------------------------------------
#define BM   128
#define BN   128
#define BKS  64
#define LDA  72
#define PAD_IVL 4   // pad after 2^(4+1)=32 DWORDs (=128 B = one 64-elem row)
#define PAD_AMT 3   // pad 4 DWORDs (=16 B)

template<bool OUT_F32>
__global__ __launch_bounds__(256)
void gemm_bf16_tn(const __bf16* __restrict__ A, const __bf16* __restrict__ Bt,
                  const float* __restrict__ bias, void* __restrict__ Cout,
                  int M, int N, int K)
{
    __shared__ __attribute__((aligned(16))) __bf16 As[2][BM * LDA];
    __shared__ __attribute__((aligned(16))) __bf16 Bs[2][BN * LDA];

    const int tid  = threadIdx.x;
    const int lane = tid & 31;
    const int wave = tid >> 5;
    const int wm   = wave >> 2;      // 0..1
    const int wn   = wave & 3;       // 0..3
    const int nh   = lane & 15;
    const int kh   = lane >> 4;
    const long bm0 = (long)blockIdx.y * BM;
    const long bn0 = (long)blockIdx.x * BN;

    const v8f vzero = {0.f,0.f,0.f,0.f,0.f,0.f,0.f,0.f};
    v8f acc[4][2];
    #pragma unroll
    for (int i = 0; i < 4; ++i)
        #pragma unroll
        for (int j = 0; j < 2; ++j) acc[i][j] = vzero;

    // prologue: wave 0 issues TDM loads for k0 = 0 into buffer 0
    if (wave == 0) {
        tdm_load_tile_bf16(lds_off_u32(&As[0][0]), A  + bm0 * K,
                           BKS, BM, (long)K, PAD_IVL, PAD_AMT);
        tdm_load_tile_bf16(lds_off_u32(&Bs[0][0]), Bt + bn0 * K,
                           BKS, BN, (long)K, PAD_IVL, PAD_AMT);
    }

    int buf = 0;
    for (int k0 = 0; k0 < K; k0 += BKS, buf ^= 1) {
        if (wave == 0) {
            __builtin_amdgcn_s_wait_tensorcnt(0);
            asm volatile("" ::: "memory");   // TDM wrote LDS behind the compiler's back
        }
        __syncthreads();
        // issue next tiles into the other buffer while computing this one
        if (wave == 0 && k0 + BKS < K) {
            tdm_load_tile_bf16(lds_off_u32(&As[buf ^ 1][0]),
                               A + bm0 * K + (k0 + BKS),
                               BKS, BM, (long)K, PAD_IVL, PAD_AMT);
            tdm_load_tile_bf16(lds_off_u32(&Bs[buf ^ 1][0]),
                               Bt + bn0 * K + (k0 + BKS),
                               BKS, BN, (long)K, PAD_IVL, PAD_AMT);
        }

        const __bf16* Ab = &As[buf][0];
        const __bf16* Bb = &Bs[buf][0];
        #pragma unroll
        for (int kc = 0; kc < 2; ++kc) {
            v16bf afr[4], bfr[2];
            #pragma unroll
            for (int i = 0; i < 4; ++i) {
                // A layout: k = {kh*8..+7, 16+kh*8..+7} within chunk
                const __bf16* p = &Ab[(wm * 64 + i * 16 + nh) * LDA + kc * 32 + kh * 8];
                afr[i] = ld_frag16(p, p + 16);
            }
            #pragma unroll
            for (int j = 0; j < 2; ++j) {
                // B layout: k = {kh*8..+7, 16+kh*8..+7} within chunk
                const __bf16* p = &Bb[(wn * 32 + j * 16 + nh) * LDA + kc * 32 + kh * 8];
                bfr[j] = ld_frag16(p, p + 16);
            }
            #pragma unroll
            for (int i = 0; i < 4; ++i)
                #pragma unroll
                for (int j = 0; j < 2; ++j)
                    acc[i][j] = wmma_bf16(afr[i], bfr[j], acc[i][j]);
        }
        __syncthreads();
    }

    // epilogue: C layout row = r + 8*kh, col = nh within each 16x16 tile
    #pragma unroll
    for (int i = 0; i < 4; ++i) {
        #pragma unroll
        for (int j = 0; j < 2; ++j) {
            long row0 = bm0 + wm * 64 + i * 16;
            int  col  = (int)bn0 + wn * 32 + j * 16 + nh;
            float bv  = bias[col];
            #pragma unroll
            for (int r = 0; r < 8; ++r) {
                long row = row0 + r + 8 * kh;
                float val = acc[i][j][r] + bv;
                if (OUT_F32)
                    ((float*)Cout)[row * N + col] = val;
                else
                    ((__bf16*)Cout)[row * N + col] = (__bf16)val;
            }
        }
    }
}

// ---------------------------------------------------------------------------
// Flash attention. One block = 128 queries of one (batch, head).
// Q/K/V as [B*S, EMB] bf16; head h uses columns [h*96, h*96+96).
// ---------------------------------------------------------------------------
#define QT 128
#define KT 64
#define LQ 104
#define LK 104
#define LV 72
#define LP 64

__global__ __launch_bounds__(256)
void flash_attn_bf16(const __bf16* __restrict__ Q, const __bf16* __restrict__ Km,
                     const __bf16* __restrict__ V, __bf16* __restrict__ Ctx,
                     float scale)
{
    __shared__ __attribute__((aligned(16))) __bf16 Qs[QT * LQ];
    __shared__ __attribute__((aligned(16))) __bf16 Ks[KT * LK];
    __shared__ __attribute__((aligned(16))) __bf16 Vt[HEAD_DIM * LV];
    __shared__ __attribute__((aligned(16))) __bf16 Ps[8 * 16 * LP];

    const int tid  = threadIdx.x;
    const int lane = tid & 31;
    const int wave = tid >> 5;
    const int nh   = lane & 15;
    const int kh   = lane >> 4;

    const int bh = blockIdx.y;
    const int b  = bh >> 3;
    const int h  = bh & 7;
    const int q0 = blockIdx.x * QT;

    const long rs = EMB;
    const long head_off = (long)h * HEAD_DIM;
    const __bf16* qbase = Q  + ((long)b * SEQ + q0) * rs + head_off;
    const __bf16* kbase = Km + ((long)b * SEQ) * rs + head_off;
    const __bf16* vbase = V  + ((long)b * SEQ) * rs + head_off;

    #pragma unroll
    for (int t = 0; t < 6; ++t) {
        int idx = tid + t * 256;
        int r = idx / 12, c = (idx % 12) * 8;
        *(uint4*)&Qs[r * LQ + c] = *(const uint4*)&qbase[(long)r * rs + c];
    }

    const v8f vzero = {0.f,0.f,0.f,0.f,0.f,0.f,0.f,0.f};
    v8f o[6];
    #pragma unroll
    for (int d = 0; d < 6; ++d) o[d] = vzero;
    float m_run[8], l_run[8];
    #pragma unroll
    for (int r = 0; r < 8; ++r) { m_run[r] = -1e30f; l_run[r] = 0.f; }

    __bf16* pw = &Ps[wave * 16 * LP];

    for (int kt = 0; kt < SEQ / KT; ++kt) {
        const __bf16* kb = kbase + (long)(kt * KT) * rs;
        const __bf16* vb = vbase + (long)(kt * KT) * rs;
        #pragma unroll
        for (int t = 0; t < 3; ++t) {
            int idx = tid + t * 256;
            int r = idx / 12, c = (idx % 12) * 8;
            *(uint4*)&Ks[r * LK + c] = *(const uint4*)&kb[(long)r * rs + c];
        }
        #pragma unroll
        for (int t = 0; t < 3; ++t) {
            int idx = tid + t * 256;
            int r = idx / 12, c = (idx % 12) * 8;     // key=r, d=c..c+7
            uint4 raw = *(const uint4*)&vb[(long)r * rs + c];
            const __bf16* e = (const __bf16*)&raw;
            #pragma unroll
            for (int j = 0; j < 8; ++j)
                Vt[(c + j) * LV + r] = e[j];
        }
        // prefetch next key tile into L2
        if (kt + 1 < SEQ / KT) {
            __builtin_prefetch(kb + (long)KT * rs + (tid % 12) * 8
                               + (long)(tid / 12) * rs, 0, 0);
        }
        __syncthreads();

        // ---- S = Q * K^T ----
        v8f sc[4];
        #pragma unroll
        for (int t = 0; t < 4; ++t) sc[t] = vzero;
        #pragma unroll
        for (int c = 0; c < 3; ++c) {
            const __bf16* ap = &Qs[(wave * 16 + nh) * LQ + c * 32 + kh * 8];
            v16bf afr = ld_frag16(ap, ap + 16);
            #pragma unroll
            for (int t = 0; t < 4; ++t) {
                const __bf16* bp = &Ks[(t * 16 + nh) * LK + c * 32 + kh * 8];
                v16bf bfr = ld_frag16(bp, bp + 16);
                sc[t] = wmma_bf16(afr, bfr, sc[t]);
            }
        }

        // ---- online softmax (row r+8*kh lives in component r of this half) ----
        float alpha[8];
        #pragma unroll
        for (int r = 0; r < 8; ++r) {
            float a0 = sc[0][r] * scale, a1 = sc[1][r] * scale;
            float a2 = sc[2][r] * scale, a3 = sc[3][r] * scale;
            sc[0][r] = a0; sc[1][r] = a1; sc[2][r] = a2; sc[3][r] = a3;
            float mx = fmaxf(fmaxf(a0, a1), fmaxf(a2, a3));
            #pragma unroll
            for (int off = 8; off >= 1; off >>= 1)
                mx = fmaxf(mx, __shfl_xor(mx, off, 16));
            float mn = fmaxf(m_run[r], mx);
            alpha[r] = __expf(m_run[r] - mn);
            m_run[r] = mn;

            float rsum = 0.f;
            #pragma unroll
            for (int t = 0; t < 4; ++t) {
                float p = __expf(sc[t][r] - mn);
                sc[t][r] = p;
                rsum += p;
            }
            #pragma unroll
            for (int off = 8; off >= 1; off >>= 1)
                rsum += __shfl_xor(rsum, off, 16);
            l_run[r] = l_run[r] * alpha[r] + rsum;
        }
        #pragma unroll
        for (int d = 0; d < 6; ++d)
            #pragma unroll
            for (int r = 0; r < 8; ++r) o[d][r] *= alpha[r];

        // ---- P (C-layout) -> per-wave LDS ----
        #pragma unroll
        for (int t = 0; t < 4; ++t)
            #pragma unroll
            for (int r = 0; r < 8; ++r)
                pw[(r + 8 * kh) * LP + t * 16 + nh] = (__bf16)sc[t][r];

        // ---- O += P * V ----
        #pragma unroll
        for (int c2 = 0; c2 < 2; ++c2) {
            const __bf16* ap = &pw[nh * LP + c2 * 32 + kh * 8];
            v16bf afr = ld_frag16(ap, ap + 16);
            #pragma unroll
            for (int d = 0; d < 6; ++d) {
                const __bf16* bp = &Vt[(d * 16 + nh) * LV + c2 * 32 + kh * 8];
                v16bf bfr = ld_frag16(bp, bp + 16);
                o[d] = wmma_bf16(afr, bfr, o[d]);
            }
        }
        __syncthreads();
    }

    __bf16* cb = Ctx + ((long)b * SEQ + q0 + wave * 16) * rs + head_off;
    #pragma unroll
    for (int d = 0; d < 6; ++d) {
        #pragma unroll
        for (int r = 0; r < 8; ++r) {
            int row = r + 8 * kh;
            float val = o[d][r] / l_run[r];
            cb[(long)row * rs + d * 16 + nh] = (__bf16)val;
        }
    }
}

// ---------------------------------------------------------------------------
// Host launcher
// ---------------------------------------------------------------------------
extern "C" void kernel_launch(void* const* d_in, const int* in_sizes, int n_in,
                              void* d_out, int out_size, void* d_ws, size_t ws_size,
                              hipStream_t stream) {
    const float* x  = (const float*)d_in[0];
    const float* wq = (const float*)d_in[1];
    const float* bq = (const float*)d_in[2];
    const float* wk = (const float*)d_in[3];
    const float* bk = (const float*)d_in[4];
    const float* wv = (const float*)d_in[5];
    const float* bv = (const float*)d_in[6];
    const float* wo = (const float*)d_in[7];
    const float* bo = (const float*)d_in[8];
    float* out = (float*)d_out;

    const int M = BATCH * SEQ;
    const int E = EMB;
    const float scale = 1.0f / sqrtf((float)HEAD_DIM);

    char* ws = (char*)d_ws;
    size_t off = 0;
    auto carve = [&](size_t bytes) {
        void* p = ws + off;
        off = (off + bytes + 255) & ~(size_t)255;
        return p;
    };
    __bf16* xb  = (__bf16*)carve((size_t)M * E * 2);
    __bf16* wqt = (__bf16*)carve((size_t)E * E * 2);   // transposed [N][K]
    __bf16* wkt = (__bf16*)carve((size_t)E * E * 2);
    __bf16* wvt = (__bf16*)carve((size_t)E * E * 2);
    __bf16* wot = (__bf16*)carve((size_t)E * E * 2);
    __bf16* qb  = (__bf16*)carve((size_t)M * E * 2);
    __bf16* kb  = (__bf16*)carve((size_t)M * E * 2);
    __bf16* vb2 = (__bf16*)carve((size_t)M * E * 2);
    __bf16* ctx = (__bf16*)carve((size_t)M * E * 2);

    // 1) conversions (weights transposed so GEMM B-operand is K-contiguous)
    {
        int n = M * E;
        cvt_f32_to_bf16<<<n / 1024, 256, 0, stream>>>(x, xb, n);
        dim3 tg(E / 32, E / 32);
        cvt_transpose_bf16<<<tg, 256, 0, stream>>>(wq, wqt, E, E);
        cvt_transpose_bf16<<<tg, 256, 0, stream>>>(wk, wkt, E, E);
        cvt_transpose_bf16<<<tg, 256, 0, stream>>>(wv, wvt, E, E);
        cvt_transpose_bf16<<<tg, 256, 0, stream>>>(wo, wot, E, E);
    }

    // 2) Q/K/V projections (bf16 out)
    dim3 ggrid(E / BN, M / BM);
    gemm_bf16_tn<false><<<ggrid, 256, 0, stream>>>(xb, wqt, bq, qb,  M, E, E);
    gemm_bf16_tn<false><<<ggrid, 256, 0, stream>>>(xb, wkt, bk, kb,  M, E, E);
    gemm_bf16_tn<false><<<ggrid, 256, 0, stream>>>(xb, wvt, bv, vb2, M, E, E);

    // 3) flash attention
    dim3 agrid(SEQ / QT, BATCH * HEADS);
    flash_attn_bf16<<<agrid, 256, 0, stream>>>(qb, kb, vb2, ctx, scale);

    // 4) output projection (fp32 out, + bias)
    gemm_bf16_tn<true><<<ggrid, 256, 0, stream>>>(ctx, wot, bo, out, M, E, E);
}
